// DualAttention_33947421508261
// MI455X (gfx1250) — compile-verified, matrix operand order checked by reading, not run on hardware
//
#include <hip/hip_runtime.h>

// ---------------------------------------------------------------------------
// Problem constants (B,S,E,H) = (4,1024,1024,16), D=64, R=32
// ---------------------------------------------------------------------------
#define B_ 4
#define S_ 1024
#define E_ 1024
#define H_ 16
#define D_ 64
#define R_ 32
#define SCALE_ 0.125f   // D^-0.5

typedef __attribute__((ext_vector_type(16))) __bf16 v16bf;
typedef __attribute__((ext_vector_type(8)))  float  v8f;

// ---------------------------------------------------------------------------
// Helpers
// ---------------------------------------------------------------------------
static __device__ __forceinline__ unsigned int pkbf(float a, float b) {
  union { __bf16 h[2]; unsigned int u; } r;
  r.h[0] = (__bf16)a; r.h[1] = (__bf16)b;
  return r.u;
}
static __device__ __forceinline__ unsigned short f2bf(float f) {
  union { __bf16 h; unsigned short u; } r; r.h = (__bf16)f; return r.u;
}
static __device__ __forceinline__ float bf2f(unsigned short u) {
  union { unsigned short u; __bf16 h; } r; r.u = u; return (float)r.h;
}

// 16x32 bf16 fragment from a row-major tile (CDNA5 16-bit A layout):
// lane holds row (lane&15); K chunks {c..c+7, c+16..c+23}, c=(lane>>4)*8.
static __device__ __forceinline__ v16bf ld_frag(const unsigned short* p, int c) {
  union { uint4 u[2]; v16bf v; } r;
  r.u[0] = *(const uint4*)(p + c);
  r.u[1] = *(const uint4*)(p + c + 16);
  return r.v;
}

static __device__ __forceinline__ v8f wmma_bf16(v16bf a, v16bf b, v8f cacc) {
  return __builtin_amdgcn_wmma_f32_16x16x32_bf16(false, a, false, b,
                                                 (short)0, cacc, false, false);
}

// Async DMA: 16 bytes global -> LDS, tracked by ASYNCcnt (no VGPR transit).
static __device__ __forceinline__ void async_cp16(unsigned lds, const unsigned short* g) {
  asm volatile("global_load_async_to_lds_b128 %0, %1, off"
               :: "v"(lds), "v"(g) : "memory");
}
static __device__ __forceinline__ void wait_async0() {
  asm volatile("s_wait_asynccnt 0x0" ::: "memory");
}

// ---------------------------------------------------------------------------
// fp32 -> bf16 bulk conversion (n multiple of 4)
// ---------------------------------------------------------------------------
__global__ void f32_to_bf16_kernel(const float* __restrict__ in,
                                   unsigned short* __restrict__ out, int n) {
  int i = (blockIdx.x * blockDim.x + threadIdx.x) * 4;
  if (i < n) {
    float4 f = *(const float4*)(in + i);
    uint2 p; p.x = pkbf(f.x, f.y); p.y = pkbf(f.z, f.w);
    *(uint2*)(out + i) = p;
  }
}

// ---------------------------------------------------------------------------
// GEMM: out[M,N] = X[M,K](bf16) * W[N,K](bf16)^T + bias[N]
// OUT_MODE 0: fp32 out [M,N].  OUT_MODE 1: bf16 out in head layout [B,H,S,D].
// 256 thr = 8 waves; block tile 128x128; K-step 64; wave tile 64x32.
// Double-buffered LDS filled by async-to-LDS DMA; one barrier per K-step.
// ---------------------------------------------------------------------------
template <int OUT_MODE>
__global__ __launch_bounds__(256)
void gemm_bf16_async_kernel(const unsigned short* __restrict__ X,
                            const unsigned short* __restrict__ W,
                            const float* __restrict__ bias, void* __restrict__ outp,
                            int M, int N, int K) {
  __shared__ __align__(16) unsigned short sA[2][128 * 72];  // stride 72 (pad)
  __shared__ __align__(16) unsigned short sB[2][128 * 72];

  const int t = threadIdx.x;
  const int lane = t & 31;
  const int wv = t >> 5;
  const int wm = wv >> 2;      // 0..1
  const int wn = wv & 3;       // 0..3
  const int lrow = lane & 15;
  const int halfsel = lane >> 4;
  const int c = halfsel * 8;
  const int m0 = blockIdx.y * 128;
  const int n0 = blockIdx.x * 128;
  const int srow = t >> 1;     // 0..127
  const int shalf = t & 1;     // 32-elem halves of the 64-wide K slab

  const unsigned baseA = (unsigned)(size_t)&sA[0][0];
  const unsigned baseB = (unsigned)(size_t)&sB[0][0];
  const unsigned tileBytes = 128u * 72u * 2u;
  const unsigned ldsoff = (unsigned)(srow * 72 + shalf * 32) * 2u;

  v8f acc[4][2];
#pragma unroll
  for (int i = 0; i < 4; i++)
#pragma unroll
    for (int j = 0; j < 2; j++) {
      v8f z = {0.f, 0.f, 0.f, 0.f, 0.f, 0.f, 0.f, 0.f};
      acc[i][j] = z;
    }

  auto stage = [&](int buf, int kk) {
    const unsigned short* gA = X + (size_t)(m0 + srow) * K + kk + shalf * 32;
    const unsigned short* gB = W + (size_t)(n0 + srow) * K + kk + shalf * 32;
    unsigned la = baseA + (unsigned)buf * tileBytes + ldsoff;
    unsigned lb = baseB + (unsigned)buf * tileBytes + ldsoff;
#pragma unroll
    for (int j = 0; j < 4; j++) {
      async_cp16(la + j * 16, gA + j * 8);
      async_cp16(lb + j * 16, gB + j * 8);
    }
  };

  stage(0, 0);
  int p = 0;
  for (int kk = 0; kk < K; kk += 64) {
    wait_async0();         // own async writes landed
    __syncthreads();       // everyone's async writes visible
    if (kk + 64 < K) stage(p ^ 1, kk + 64);   // prefetch next tile (overlaps)
#pragma unroll
    for (int kc = 0; kc < 2; kc++) {
      v16bf bf0 = ld_frag(&sB[p][(wn * 32 + 0 * 16 + lrow) * 72 + kc * 32], c);
      v16bf bf1 = ld_frag(&sB[p][(wn * 32 + 1 * 16 + lrow) * 72 + kc * 32], c);
#pragma unroll
      for (int mi = 0; mi < 4; mi++) {
        v16bf af = ld_frag(&sA[p][(wm * 64 + mi * 16 + lrow) * 72 + kc * 32], c);
        acc[mi][0] = wmma_bf16(af, bf0, acc[mi][0]);
        acc[mi][1] = wmma_bf16(af, bf1, acc[mi][1]);
      }
    }
    p ^= 1;
  }

  // Epilogue. C layout: VGPR i -> row (i + 8*halfsel), col = lane&15 per tile.
#pragma unroll
  for (int mi = 0; mi < 4; mi++) {
#pragma unroll
    for (int ni = 0; ni < 2; ni++) {
      const int col = n0 + wn * 32 + ni * 16 + lrow;
      const float bv = bias[col];
#pragma unroll
      for (int i = 0; i < 8; i++) {
        const int row = m0 + wm * 64 + mi * 16 + i + (halfsel << 3);
        float v = acc[mi][ni][i] + bv;
        if (OUT_MODE == 0) {
          ((float*)outp)[(size_t)row * N + col] = v;
        } else {
          const int b = row >> 10;          // /S
          const int s = row & (S_ - 1);
          const int h = col >> 6;           // /D
          const int d = col & (D_ - 1);
          ((unsigned short*)outp)[(((size_t)b * H_ + h) * S_ + s) * D_ + d] = f2bf(v);
        }
      }
    }
  }
}

// ---------------------------------------------------------------------------
// RoPE in-place on bf16 [B,H,S,D]; rotary over first R=32 dims per head.
// ---------------------------------------------------------------------------
__global__ void rope_kernel(unsigned short* __restrict__ X,
                            const float* __restrict__ cosp,
                            const float* __restrict__ sinp) {
  int idx = blockIdx.x * blockDim.x + threadIdx.x;  // B*H*S*16 threads
  int d = idx & 15;
  int s = (idx >> 4) & (S_ - 1);
  int bh = idx >> 14;
  size_t off = ((size_t)bh * S_ + s) * D_;
  float x0 = bf2f(X[off + d]);
  float x1 = bf2f(X[off + d + 16]);
  float c0 = cosp[s * R_ + d],      s0 = sinp[s * R_ + d];
  float c1 = cosp[s * R_ + d + 16], s1 = sinp[s * R_ + d + 16];
  X[off + d]      = f2bf(x0 * c0 - x1 * s0);
  X[off + d + 16] = f2bf(x1 * c1 + x0 * s1);
}

// ---------------------------------------------------------------------------
// Mask -> additive bias
// ---------------------------------------------------------------------------
__global__ void maskbias_kernel(const int* __restrict__ m1, const int* __restrict__ m2,
                                float* __restrict__ mb1, float* __restrict__ mb2, int n) {
  int i = blockIdx.x * blockDim.x + threadIdx.x;
  if (i < n) {
    mb1[i] = m1[i] ? 0.f : -1e30f;
    mb2[i] = m2[i] ? 0.f : -1e30f;
  }
}

// ---------------------------------------------------------------------------
// Flash attention: grid (S/128, B*H), block 256 = 8 waves.
// Wave owns 16 query rows; 64-key K/V tiles streamed through LDS
// (K tile via async-to-LDS DMA, V transposed through registers).
// ---------------------------------------------------------------------------
__global__ __launch_bounds__(256)
void attn_kernel(const unsigned short* __restrict__ Q,
                 const unsigned short* __restrict__ Kb,
                 const unsigned short* __restrict__ Vb,
                 const float* __restrict__ mbias,
                 float* __restrict__ att, int accum) {
  __shared__ __align__(16) unsigned short sK[64 * 72];       // [key][d]
  __shared__ __align__(16) unsigned short sVt[64 * 72];      // [d][key]
  __shared__ __align__(16) unsigned short sP[8 * 16 * 72];   // per-wave P scratch
  __shared__ float sMb[64];

  const int t = threadIdx.x;
  const int lane = t & 31;
  const int wv = t >> 5;
  const int lrow = lane & 15;
  const int c = (lane >> 4) * 8;
  const int bh = blockIdx.y;
  const int b = bh >> 4;   // /H
  const int h = bh & 15;
  const int q0 = blockIdx.x * 128 + wv * 16;

  const unsigned baseK = (unsigned)(size_t)&sK[0];

  // Q fragments for this wave's 16 rows (two 16x32 frags over D=64).
  const unsigned short* qrow = Q + ((size_t)bh * S_ + (q0 + lrow)) * D_;
  v16bf qf0 = ld_frag(qrow, c);
  v16bf qf1 = ld_frag(qrow + 32, c);

  v8f o[4];
#pragma unroll
  for (int i = 0; i < 4; i++) { v8f z = {0,0,0,0,0,0,0,0}; o[i] = z; }
  float rm[8], rs[8];
#pragma unroll
  for (int i = 0; i < 8; i++) { rm[i] = -1e30f; rs[i] = 0.f; }

  unsigned short* sPw = &sP[wv * 16 * 72];
  const int skey = t >> 2, sq4 = t & 3;

  for (int kk0 = 0; kk0 < S_; kk0 += 64) {
    // ---- async-stage K tile ----
    {
      const unsigned short* ks = Kb + ((size_t)bh * S_ + kk0 + skey) * D_ + sq4 * 16;
      unsigned lk = baseK + (unsigned)(skey * 72 + sq4 * 16) * 2u;
      async_cp16(lk, ks);
      async_cp16(lk + 16, ks + 8);
    }
    // ---- stage V transposed (register transpose) + mask bias ----
    {
      const unsigned short* vs = Vb + ((size_t)bh * S_ + kk0 + skey) * D_ + sq4 * 16;
      union { uint4 u[2]; unsigned short s[16]; } vvv;
      vvv.u[0] = ((const uint4*)vs)[0];
      vvv.u[1] = ((const uint4*)vs)[1];
#pragma unroll
      for (int e = 0; e < 16; e++)
        sVt[(sq4 * 16 + e) * 72 + skey] = vvv.s[e];
      if (t < 64) sMb[t] = mbias[(size_t)b * S_ + kk0 + t];
    }
    wait_async0();
    __syncthreads();

    // ---- scores: 4 key-tiles of 16, contraction over D=64 ----
    float sc[4][8];
#pragma unroll
    for (int kt = 0; kt < 4; kt++) {
      v16bf kf0 = ld_frag(&sK[(kt * 16 + lrow) * 72], c);
      v16bf kf1 = ld_frag(&sK[(kt * 16 + lrow) * 72 + 32], c);
      v8f a = {0,0,0,0,0,0,0,0};
      a = wmma_bf16(qf0, kf0, a);
      a = wmma_bf16(qf1, kf1, a);
      float mb = sMb[kt * 16 + lrow];
#pragma unroll
      for (int i = 0; i < 8; i++) sc[kt][i] = a[i] * SCALE_ + mb;
    }

    // ---- online softmax (row reductions across 16-lane halves) ----
#pragma unroll
    for (int i = 0; i < 8; i++) {
      float m = fmaxf(fmaxf(sc[0][i], sc[1][i]), fmaxf(sc[2][i], sc[3][i]));
      for (int off = 1; off < 16; off <<= 1) m = fmaxf(m, __shfl_xor(m, off, 32));
      float nm = fmaxf(rm[i], m);
      float rescale = __expf(rm[i] - nm);
      rm[i] = nm;
      rs[i] *= rescale;
#pragma unroll
      for (int ni = 0; ni < 4; ni++) o[ni][i] *= rescale;
      float psum = 0.f;
#pragma unroll
      for (int kt = 0; kt < 4; kt++) {
        float p = __expf(sc[kt][i] - nm);
        sc[kt][i] = p;
        psum += p;
      }
      for (int off = 1; off < 16; off <<= 1) psum += __shfl_xor(psum, off, 32);
      rs[i] += psum;
    }

    // ---- C-layout -> A-layout for P via per-wave LDS scratch ----
#pragma unroll
    for (int kt = 0; kt < 4; kt++) {
#pragma unroll
      for (int i = 0; i < 8; i++) {
        int row = i + ((lane >> 4) << 3);
        sPw[row * 72 + kt * 16 + lrow] = f2bf(sc[kt][i]);
      }
    }
    v16bf pf0 = ld_frag(&sPw[lrow * 72], c);
    v16bf pf1 = ld_frag(&sPw[lrow * 72 + 32], c);

    // ---- O += P @ V ----
#pragma unroll
    for (int ni = 0; ni < 4; ni++) {
      v16bf vf0 = ld_frag(&sVt[(ni * 16 + lrow) * 72], c);
      v16bf vf1 = ld_frag(&sVt[(ni * 16 + lrow) * 72 + 32], c);
      o[ni] = wmma_bf16(pf0, vf0, o[ni]);
      o[ni] = wmma_bf16(pf1, vf1, o[ni]);
    }
    __syncthreads();
  }

  // ---- normalize and write/accumulate into att [B,S,E] ----
#pragma unroll
  for (int i = 0; i < 8; i++) {
    float inv = 1.0f / rs[i];
    int s = q0 + i + ((lane >> 4) << 3);
#pragma unroll
    for (int ni = 0; ni < 4; ni++) {
      size_t idx = ((size_t)b * S_ + s) * E_ + h * D_ + ni * 16 + lrow;
      float v = o[ni][i] * inv;
      if (accum) att[idx] += v; else att[idx] = v;
    }
  }
}

// ---------------------------------------------------------------------------
// Launcher
// ---------------------------------------------------------------------------
extern "C" void kernel_launch(void* const* d_in, const int* in_sizes, int n_in,
                              void* d_out, int out_size, void* d_ws, size_t ws_size,
                              hipStream_t stream) {
  (void)in_sizes; (void)n_in; (void)out_size; (void)ws_size;
  const float* x1   = (const float*)d_in[0];
  const float* x2   = (const float*)d_in[1];
  const int*   pm1  = (const int*)d_in[2];
  const int*   pm2  = (const int*)d_in[3];
  const float* cos1 = (const float*)d_in[4];
  const float* sin1 = (const float*)d_in[5];
  const float* cos2 = (const float*)d_in[6];
  const float* sin2 = (const float*)d_in[7];
  const float* Wf[8]; const float* bf[8];
  for (int i = 0; i < 8; i++) { Wf[i] = (const float*)d_in[8 + 2 * i]; bf[i] = (const float*)d_in[9 + 2 * i]; }
  // name order: 0=q1 1=k2 2=v2 3=q2 4=k1 5=v1 6=o1 7=o2

  char* ws = (char*)d_ws;
  const size_t QKV   = (size_t)B_ * H_ * S_ * D_ * 2;   // 8 MB bf16
  const size_t ATTSZ = (size_t)B_ * S_ * E_ * 4;        // 16 MB fp32
  const size_t XBF   = (size_t)B_ * S_ * E_ * 2;        // 8 MB bf16
  const size_t WBF   = (size_t)E_ * E_ * 2;             // 2 MB bf16
  unsigned short* q1b = (unsigned short*)(ws + 0 * QKV);
  unsigned short* k1b = (unsigned short*)(ws + 1 * QKV);
  unsigned short* v1b = (unsigned short*)(ws + 2 * QKV);
  unsigned short* q2b = (unsigned short*)(ws + 3 * QKV);
  unsigned short* k2b = (unsigned short*)(ws + 4 * QKV);
  unsigned short* v2b = (unsigned short*)(ws + 5 * QKV);
  float* att1 = (float*)(ws + 6 * QKV);
  float* att2 = (float*)(ws + 6 * QKV + ATTSZ);
  float* mb1  = (float*)(ws + 6 * QKV + 2 * ATTSZ);
  float* mb2  = mb1 + B_ * S_;
  char* p = ws + 6 * QKV + 2 * ATTSZ + (1 << 20);
  unsigned short* x1b = (unsigned short*)(p);            // later reused as att1 bf16
  unsigned short* x2b = (unsigned short*)(p + XBF);      // later reused as att2 bf16
  unsigned short* Wb[8];
  for (int i = 0; i < 8; i++) Wb[i] = (unsigned short*)(p + 2 * XBF + i * WBF);

  const int M = B_ * S_, N = E_, K = E_;
  const int nX = B_ * S_ * E_, nW = E_ * E_;

  // fp32 -> bf16 operand conversion
  f32_to_bf16_kernel<<<nX / 4 / 256, 256, 0, stream>>>(x1, x1b, nX);
  f32_to_bf16_kernel<<<nX / 4 / 256, 256, 0, stream>>>(x2, x2b, nX);
  for (int i = 0; i < 8; i++)
    f32_to_bf16_kernel<<<nW / 4 / 256, 256, 0, stream>>>(Wf[i], Wb[i], nW);

  maskbias_kernel<<<(B_ * S_ + 255) / 256, 256, 0, stream>>>(pm1, pm2, mb1, mb2, B_ * S_);

  dim3 ggrid(N / 128, M / 128);   // (8, 32)
  // QKV projections (bf16 head-layout outputs)
  gemm_bf16_async_kernel<1><<<ggrid, 256, 0, stream>>>(x1b, Wb[0], bf[0], q1b, M, N, K); // q1
  gemm_bf16_async_kernel<1><<<ggrid, 256, 0, stream>>>(x1b, Wb[4], bf[4], k1b, M, N, K); // k1
  gemm_bf16_async_kernel<1><<<ggrid, 256, 0, stream>>>(x1b, Wb[5], bf[5], v1b, M, N, K); // v1
  gemm_bf16_async_kernel<1><<<ggrid, 256, 0, stream>>>(x2b, Wb[3], bf[3], q2b, M, N, K); // q2
  gemm_bf16_async_kernel<1><<<ggrid, 256, 0, stream>>>(x2b, Wb[1], bf[1], k2b, M, N, K); // k2
  gemm_bf16_async_kernel<1><<<ggrid, 256, 0, stream>>>(x2b, Wb[2], bf[2], v2b, M, N, K); // v2

  // RoPE on q1/k1 (cos1/sin1), q2/k2 (cos2/sin2)
  const int rthreads = B_ * H_ * S_ * 16;
  rope_kernel<<<rthreads / 256, 256, 0, stream>>>(q1b, cos1, sin1);
  rope_kernel<<<rthreads / 256, 256, 0, stream>>>(k1b, cos1, sin1);
  rope_kernel<<<rthreads / 256, 256, 0, stream>>>(q2b, cos2, sin2);
  rope_kernel<<<rthreads / 256, 256, 0, stream>>>(k2b, cos2, sin2);

  // attended_1 = attn(q1,k2,v2,m2) + attn(q1,k1,v1,m1); attended_2 symmetric
  dim3 agrid(S_ / 128, B_ * H_);  // (8, 64)
  attn_kernel<<<agrid, 256, 0, stream>>>(q1b, k2b, v2b, mb2, att1, 0);
  attn_kernel<<<agrid, 256, 0, stream>>>(q1b, k1b, v1b, mb1, att1, 1);
  attn_kernel<<<agrid, 256, 0, stream>>>(q2b, k1b, v1b, mb1, att2, 0);
  attn_kernel<<<agrid, 256, 0, stream>>>(q2b, k2b, v2b, mb2, att2, 1);

  // att fp32 -> bf16 (reuse x1b/x2b regions, x inputs are dead now)
  f32_to_bf16_kernel<<<nX / 4 / 256, 256, 0, stream>>>(att1, x1b, nX);
  f32_to_bf16_kernel<<<nX / 4 / 256, 256, 0, stream>>>(att2, x2b, nX);

  // Output projections -> d_out (fp32, tuple concat: out1 then out2)
  float* out1 = (float*)d_out;
  float* out2 = out1 + (size_t)B_ * S_ * E_;
  gemm_bf16_async_kernel<0><<<ggrid, 256, 0, stream>>>(x1b, Wb[6], bf[6], out1, M, N, K); // o1
  gemm_bf16_async_kernel<0><<<ggrid, 256, 0, stream>>>(x2b, Wb[7], bf[7], out2, M, N, K); // o2
}
